// _GNNBlock_3693671874573
// MI455X (gfx1250) — matrix operand order repaired and checked by reading, not compiled, not run on hardware
//
#include <hip/hip_runtime.h>

// ---------------------------------------------------------------------------
// GNN block for MI455X (gfx1250, wave32, WMMA).
// Edge MLP + scatter-mean + node MLP, all matmuls on v_wmma_f32_16x16x32_f16.
// Edge kernel: M=32 rows/wave (B-fragment reuse x2), W1 staged in LDS,
// persistent blocks to amortize the staging, V pre-converted to fp16 so the
// S/R gathers are direct v8h loads (no per-use cvt VALU, half the L2 traffic).
// ---------------------------------------------------------------------------

typedef __attribute__((ext_vector_type(16))) _Float16 v16h;
typedef __attribute__((ext_vector_type(8)))  _Float16 v8h;
typedef __attribute__((ext_vector_type(8)))  float    v8f;

static constexpr int B_  = 2;
static constexpr int N_  = 10000;
static constexpr int NE_ = 320000;
static constexpr int ND_ = 128;
static constexpr int ED_ = 128;

// --- edge kernel geometry: M=32 per wave, persistent blocks ---
static constexpr int WPB_E   = 4;                      // waves per block
static constexpr int EDGE_W32 = (B_ * NE_) / 32;       // 20000 wave-tiles of 32 edges
static constexpr int NBLK_E  = 500;                    // persistent blocks
static constexpr int ITER_E  = EDGE_W32 / (NBLK_E * WPB_E);  // 10
static_assert(NBLK_E * WPB_E * ITER_E == EDGE_W32, "edge grid exact");
static_assert((NE_ / 32) % (WPB_E * ITER_E) == 0, "edge blocks don't straddle batches");

// --- node kernel geometry: M=16 per wave ---
static constexpr int WPB_N = 2;
static constexpr int NODE_TILES = (B_ * N_) / 16;      // 1250
static_assert(NODE_TILES % WPB_N == 0, "node grid exact");
static_assert(N_ % 16 == 0, "node tiles don't straddle batches");

static constexpr int W1E_HALVES = 384 * 128;           // 49152

// Workspace layout (bytes)
static constexpr size_t AGG_OFF = 0;                                    // B*N*ED f32
static constexpr size_t CNT_OFF = AGG_OFF + (size_t)B_ * N_ * ED_ * 4;  // B*N f32
static constexpr size_t W1E_OFF = CNT_OFF + (size_t)B_ * N_ * 4;        // 384*128 halves
static constexpr size_t W2E_OFF = W1E_OFF + 384 * 128 * 2;              // 128*128 halves
static constexpr size_t W1N_OFF = W2E_OFF + 128 * 128 * 2;              // 256*128 halves
static constexpr size_t W2N_OFF = W1N_OFF + 256 * 128 * 2;              // 128*128 halves
static constexpr size_t VH_OFF  = W2N_OFF + 128 * 128 * 2;              // B*N*ND halves

// ---------------------------------------------------------------------------
// Pack a row-major f32 weight [K][128] into WMMA B-fragment order, f16.
// Fragment (kt,nt): 32 lanes x 16 halves; element (lane,e):
//   k = kt*32 + (e<8 ? (lane>>4)*8 + e : 16 + (lane>>4)*8 + (e-8)),  n = nt*16 + (lane&15)
// ---------------------------------------------------------------------------
__global__ void pack_w_kernel(const float* __restrict__ w, _Float16* __restrict__ dst,
                              int K) {
  int idx = blockIdx.x * blockDim.x + threadIdx.x;
  int total = K * 128;
  if (idx >= total) return;
  int e    = idx & 15;
  int lane = (idx >> 4) & 31;
  int nt   = (idx >> 9) & 7;
  int kt   = idx >> 12;
  int hi   = lane >> 4;
  int n = nt * 16 + (lane & 15);
  int k = kt * 32 + ((e < 8) ? (hi * 8 + e) : (16 + hi * 8 + (e - 8)));
  dst[idx] = (_Float16)w[(size_t)k * 128 + n];
}

__global__ void zero_kernel(float* __restrict__ p, int n) {
  int i = blockIdx.x * blockDim.x + threadIdx.x;
  if (i < n) p[i] = 0.0f;
}

__global__ void conv_v_kernel(const float* __restrict__ src, _Float16* __restrict__ dst,
                              int n) {
  int i = blockIdx.x * blockDim.x + threadIdx.x;
  if (i < n) dst[i] = (_Float16)src[i];
}

// ---------------------------------------------------------------------------
// Edge kernel. 1 wave = 32 edges (two 16-row A fragments share each B frag).
// W1 (384x128 f16 fragments, 96KB) staged in LDS once per persistent block.
// S/R gathers come from pre-converted fp16 V (direct v8h loads).
// ---------------------------------------------------------------------------
__global__ __launch_bounds__(WPB_E * 32) void edge_kernel(
    const _Float16* __restrict__ Vh, const float* __restrict__ E,
    const int* __restrict__ edges,
    const _Float16* __restrict__ w1, const _Float16* __restrict__ w2,
    const float* __restrict__ b1, const float* __restrict__ b2,
    const float* __restrict__ gln, const float* __restrict__ bln,
    float* __restrict__ aggSum, float* __restrict__ cnt,
    float* __restrict__ outE) {

  __shared__ __attribute__((aligned(16))) _Float16 s_W1[W1E_HALVES];    // 96 KB
  __shared__ __attribute__((aligned(16))) _Float16 s_H[WPB_E][32][136]; // ~34 KB
  __shared__ int   s_src[WPB_E][32];
  __shared__ int   s_dst[WPB_E][32];
  __shared__ float s_val[WPB_E][32];

  const int w    = threadIdx.x >> 5;
  const int lane = threadIdx.x & 31;
  const int col  = lane & 15;
  const int hi   = lane >> 4;
  const int r0   = col;        // A-fragment row, first half  (edges 0..15)
  const int r1   = 16 + col;   // A-fragment row, second half (edges 16..31)

  // ---- stage W1 fragments into LDS (6144 v8h / 128 threads = 48 iters) ----
  {
    v8h* d = (v8h*)s_W1;
    const v8h* s = (const v8h*)w1;
#pragma unroll
    for (int i = 0; i < W1E_HALVES / 8 / (WPB_E * 32); ++i)
      d[i * (WPB_E * 32) + threadIdx.x] = s[i * (WPB_E * 32) + threadIdx.x];
  }
  __syncthreads();

  for (int it = 0; it < ITER_E; ++it) {
    const int t32 = (blockIdx.x * ITER_E + it) * WPB_E + w;  // wave-tile of 32 edges
    const int b = t32 / (NE_ / 32);
    const size_t edge0 = (size_t)t32 * 32;                   // flat edge index

    // ---- phase 0: indices, validity, count atomics (one edge per lane) ----
    {
      size_t ge = edge0 + lane;
      int s = edges[ge * 2 + 0];
      int r = edges[ge * 2 + 1];
      float valid = (s >= 0 && r >= 0) ? 1.0f : 0.0f;
      int ss = min(max(s, 0), N_ - 1);
      int rr = min(max(r, 0), N_ - 1);
      s_src[w][lane] = ss;
      s_dst[w][lane] = rr;
      s_val[w][lane] = valid;
      atomicAdd(&cnt[(size_t)b * N_ + rr], 1.0f);  // invalid edges count at clamped idx
    }
    __syncthreads();

    const _Float16* srcH0 = Vh + ((size_t)b * N_ + s_src[w][r0]) * ND_;
    const _Float16* srcH1 = Vh + ((size_t)b * N_ + s_src[w][r1]) * ND_;
    const _Float16* rcvH0 = Vh + ((size_t)b * N_ + s_dst[w][r0]) * ND_;
    const _Float16* rcvH1 = Vh + ((size_t)b * N_ + s_dst[w][r1]) * ND_;
    const float*    eP0   = E + (edge0 + r0) * ED_;
    const float*    eP1   = E + (edge0 + r1) * ED_;

    // ---- phase 1: GEMM1  X(32x384) @ W1(384x128), B fragments from LDS ----
    v8f acc[2][8];
#pragma unroll
    for (int mh = 0; mh < 2; ++mh)
#pragma unroll
      for (int nt = 0; nt < 8; ++nt)
#pragma unroll
        for (int j = 0; j < 8; ++j) acc[mh][nt][j] = 0.0f;

    for (int kt = 0; kt < 12; ++kt) {
      int segk = (kt & 3) * 32;
      int k1 = segk + hi * 8;
      int k2 = segk + 16 + hi * 8;
      v16h a0, a1;
      if (kt < 8) {
        // S / R segments: fp16 gather, no conversion needed
        const _Float16* h0 = (kt < 4) ? srcH0 : rcvH0;
        const _Float16* h1 = (kt < 4) ? srcH1 : rcvH1;
        v8h x0 = *(const v8h*)(h0 + k1);
        v8h x1 = *(const v8h*)(h0 + k2);
        v8h y0 = *(const v8h*)(h1 + k1);
        v8h y1 = *(const v8h*)(h1 + k2);
#pragma unroll
        for (int j = 0; j < 8; ++j) {
          a0[j] = x0[j]; a0[8 + j] = x1[j];
          a1[j] = y0[j]; a1[8 + j] = y1[j];
        }
      } else {
        // E segment: fp32 stream, convert
        float4 u0 = *(const float4*)(eP0 + k1);
        float4 u1 = *(const float4*)(eP0 + k1 + 4);
        float4 u2 = *(const float4*)(eP0 + k2);
        float4 u3 = *(const float4*)(eP0 + k2 + 4);
        float4 q0 = *(const float4*)(eP1 + k1);
        float4 q1 = *(const float4*)(eP1 + k1 + 4);
        float4 q2 = *(const float4*)(eP1 + k2);
        float4 q3 = *(const float4*)(eP1 + k2 + 4);
        a0[0]=(_Float16)u0.x; a0[1]=(_Float16)u0.y; a0[2]=(_Float16)u0.z; a0[3]=(_Float16)u0.w;
        a0[4]=(_Float16)u1.x; a0[5]=(_Float16)u1.y; a0[6]=(_Float16)u1.z; a0[7]=(_Float16)u1.w;
        a0[8]=(_Float16)u2.x; a0[9]=(_Float16)u2.y; a0[10]=(_Float16)u2.z; a0[11]=(_Float16)u2.w;
        a0[12]=(_Float16)u3.x; a0[13]=(_Float16)u3.y; a0[14]=(_Float16)u3.z; a0[15]=(_Float16)u3.w;
        a1[0]=(_Float16)q0.x; a1[1]=(_Float16)q0.y; a1[2]=(_Float16)q0.z; a1[3]=(_Float16)q0.w;
        a1[4]=(_Float16)q1.x; a1[5]=(_Float16)q1.y; a1[6]=(_Float16)q1.z; a1[7]=(_Float16)q1.w;
        a1[8]=(_Float16)q2.x; a1[9]=(_Float16)q2.y; a1[10]=(_Float16)q2.z; a1[11]=(_Float16)q2.w;
        a1[12]=(_Float16)q3.x; a1[13]=(_Float16)q3.y; a1[14]=(_Float16)q3.z; a1[15]=(_Float16)q3.w;
      }
#pragma unroll
      for (int nt = 0; nt < 8; ++nt) {
        const _Float16* fb = s_W1 + ((size_t)((kt * 8 + nt) * 32 + lane)) * 16;
        v8h f0 = *(const v8h*)fb;
        v8h f1 = *(const v8h*)(fb + 8);
        v16h bf;
#pragma unroll
        for (int j = 0; j < 8; ++j) { bf[j] = f0[j]; bf[8 + j] = f1[j]; }
        acc[0][nt] = __builtin_amdgcn_wmma_f32_16x16x32_f16(
            false, a0, false, bf, (short)0, acc[0][nt], false, false);
        acc[1][nt] = __builtin_amdgcn_wmma_f32_16x16x32_f16(
            false, a1, false, bf, (short)0, acc[1][nt], false, false);
      }
    }

    // ---- phase 2: +b1, SiLU, stage H (f16, 32 rows) in LDS ----
    {
      float bv[8];
#pragma unroll
      for (int nt = 0; nt < 8; ++nt) bv[nt] = b1[nt * 16 + col];
#pragma unroll
      for (int mh = 0; mh < 2; ++mh)
#pragma unroll
        for (int nt = 0; nt < 8; ++nt) {
          int n = nt * 16 + col;
#pragma unroll
          for (int r = 0; r < 8; ++r) {
            float x = acc[mh][nt][r] + bv[nt];
            float h = x / (1.0f + __expf(-x));
            s_H[w][mh * 16 + hi * 8 + r][n] = (_Float16)h;
          }
        }
    }
    __syncthreads();

    // ---- phase 3: GEMM2  H(32x128) @ W2(128x128), B fragments from L2 ----
    v8f acc2[2][8];
#pragma unroll
    for (int mh = 0; mh < 2; ++mh)
#pragma unroll
      for (int nt = 0; nt < 8; ++nt)
#pragma unroll
        for (int j = 0; j < 8; ++j) acc2[mh][nt][j] = 0.0f;

#pragma unroll
    for (int kt = 0; kt < 4; ++kt) {
      int k1 = kt * 32 + hi * 8;
      int k2 = kt * 32 + 16 + hi * 8;
      v8h g0 = *(const v8h*)&s_H[w][r0][k1];
      v8h g1 = *(const v8h*)&s_H[w][r0][k2];
      v8h g2 = *(const v8h*)&s_H[w][r1][k1];
      v8h g3 = *(const v8h*)&s_H[w][r1][k2];
      v16h a0, a1;
#pragma unroll
      for (int j = 0; j < 8; ++j) {
        a0[j] = g0[j]; a0[8 + j] = g1[j];
        a1[j] = g2[j]; a1[8 + j] = g3[j];
      }
#pragma unroll
      for (int nt = 0; nt < 8; ++nt) {
        v16h bf = *(const v16h*)(w2 + ((size_t)((kt * 8 + nt) * 32 + lane)) * 16);
        acc2[0][nt] = __builtin_amdgcn_wmma_f32_16x16x32_f16(
            false, a0, false, bf, (short)0, acc2[0][nt], false, false);
        acc2[1][nt] = __builtin_amdgcn_wmma_f32_16x16x32_f16(
            false, a1, false, bf, (short)0, acc2[1][nt], false, false);
      }
    }

    // ---- phase 4: +b2, LayerNorm over 128 cols, mask, outputs ----
    {
      float b2v[8], gv[8], bev[8];
#pragma unroll
      for (int nt = 0; nt < 8; ++nt) {
        int n = nt * 16 + col;
        b2v[nt] = b2[n]; gv[nt] = gln[n]; bev[nt] = bln[n];
      }
#pragma unroll
      for (int mh = 0; mh < 2; ++mh) {
        float sum[8], sq[8];
#pragma unroll
        for (int r = 0; r < 8; ++r) { sum[r] = 0.0f; sq[r] = 0.0f; }
#pragma unroll
        for (int nt = 0; nt < 8; ++nt)
#pragma unroll
          for (int r = 0; r < 8; ++r) {
            float y = acc2[mh][nt][r] + b2v[nt];
            acc2[mh][nt][r] = y;
            sum[r] += y; sq[r] += y * y;
          }
#pragma unroll
        for (int m = 1; m < 16; m <<= 1) {
#pragma unroll
          for (int r = 0; r < 8; ++r) {
            sum[r] += __shfl_xor(sum[r], m);
            sq[r]  += __shfl_xor(sq[r], m);
          }
        }
        int   dstr[8];
        float valr[8];
#pragma unroll
        for (int r = 0; r < 8; ++r) {
          int m = mh * 16 + hi * 8 + r;
          dstr[r] = s_dst[w][m];
          valr[r] = s_val[w][m];
        }
#pragma unroll
        for (int nt = 0; nt < 8; ++nt) {
          int n = nt * 16 + col;
#pragma unroll
          for (int r = 0; r < 8; ++r) {
            float muv = sum[r] * (1.0f / 128.0f);
            float var = sq[r] * (1.0f / 128.0f) - muv * muv;
            float rsv = rsqrtf(var + 1e-5f);
            float yn = (acc2[mh][nt][r] - muv) * rsv * gv[nt] + bev[nt];
            float en = yn * valr[r];
            size_t rowIdx = edge0 + mh * 16 + hi * 8 + r;
            outE[rowIdx * ED_ + n] = E[rowIdx * ED_ + n] + en;
            atomicAdd(&aggSum[((size_t)b * N_ + dstr[r]) * ED_ + n], en);
          }
        }
      }
    }
    // per-wave LDS reuse across iterations is in-order (DS ops in order per
    // wave; same-array dynamic-index accesses cannot be reordered); the
    // __syncthreads after next iteration's phase 0 re-establishes visibility.
  }
}

// ---------------------------------------------------------------------------
// Node kernel: 1 wave = 16 nodes. X = [V | agg] (16x256) -> W1n -> SiLU ->
// W2n -> +b2 -> LayerNorm. Writes V + v_new (residual read from fp32 V).
// ---------------------------------------------------------------------------
__global__ __launch_bounds__(WPB_N * 32) void node_kernel(
    const float* __restrict__ V, const _Float16* __restrict__ Vh,
    const float* __restrict__ aggSum, const float* __restrict__ cnt,
    const _Float16* __restrict__ w1, const _Float16* __restrict__ w2,
    const float* __restrict__ b1, const float* __restrict__ b2,
    const float* __restrict__ gln, const float* __restrict__ bln,
    float* __restrict__ outV) {

  __shared__ float    s_inv[WPB_N][16];
  __shared__ __attribute__((aligned(16))) _Float16 s_H[WPB_N][16][136];

  const int w    = threadIdx.x >> 5;
  const int lane = threadIdx.x & 31;
  const int tile = blockIdx.x * WPB_N + w;   // exact grid
  const int rowA = lane & 15;
  const int col  = lane & 15;
  const int hi   = lane >> 4;

  if (lane < 16) {
    size_t gn = (size_t)tile * 16 + lane;
    float c = cnt[gn];
    s_inv[w][lane] = 1.0f / fmaxf(c, 1.0f);
  }
  __syncthreads();

  size_t gnRow = (size_t)tile * 16 + rowA;
  const _Float16* vH = Vh + gnRow * ND_;
  const float*    aP = aggSum + gnRow * ED_;
  float inv = s_inv[w][rowA];

  v8f acc[8];
#pragma unroll
  for (int nt = 0; nt < 8; ++nt)
#pragma unroll
    for (int j = 0; j < 8; ++j) acc[nt][j] = 0.0f;

  for (int kt = 0; kt < 8; ++kt) {
    int segk = (kt & 3) * 32;
    int k1 = segk + hi * 8;
    int k2 = segk + 16 + hi * 8;
    v16h a;
    if (kt < 4) {
      v8h x0 = *(const v8h*)(vH + k1);
      v8h x1 = *(const v8h*)(vH + k2);
#pragma unroll
      for (int j = 0; j < 8; ++j) { a[j] = x0[j]; a[8 + j] = x1[j]; }
    } else {
      float4 u0 = *(const float4*)(aP + k1);
      float4 u1 = *(const float4*)(aP + k1 + 4);
      float4 u2 = *(const float4*)(aP + k2);
      float4 u3 = *(const float4*)(aP + k2 + 4);
      a[0]=(_Float16)(u0.x*inv); a[1]=(_Float16)(u0.y*inv);
      a[2]=(_Float16)(u0.z*inv); a[3]=(_Float16)(u0.w*inv);
      a[4]=(_Float16)(u1.x*inv); a[5]=(_Float16)(u1.y*inv);
      a[6]=(_Float16)(u1.z*inv); a[7]=(_Float16)(u1.w*inv);
      a[8]=(_Float16)(u2.x*inv); a[9]=(_Float16)(u2.y*inv);
      a[10]=(_Float16)(u2.z*inv); a[11]=(_Float16)(u2.w*inv);
      a[12]=(_Float16)(u3.x*inv); a[13]=(_Float16)(u3.y*inv);
      a[14]=(_Float16)(u3.z*inv); a[15]=(_Float16)(u3.w*inv);
    }
#pragma unroll
    for (int nt = 0; nt < 8; ++nt) {
      v16h bf = *(const v16h*)(w1 + ((size_t)((kt * 8 + nt) * 32 + lane)) * 16);
      acc[nt] = __builtin_amdgcn_wmma_f32_16x16x32_f16(
          false, a, false, bf, (short)0, acc[nt], false, false);
    }
  }

#pragma unroll
  for (int nt = 0; nt < 8; ++nt) {
    int n = nt * 16 + col;
    float bv = b1[n];
#pragma unroll
    for (int r = 0; r < 8; ++r) {
      float x = acc[nt][r] + bv;
      float h = x / (1.0f + __expf(-x));
      s_H[w][hi * 8 + r][n] = (_Float16)h;
    }
  }
  __syncthreads();

  v8f acc2[8];
#pragma unroll
  for (int nt = 0; nt < 8; ++nt)
#pragma unroll
    for (int j = 0; j < 8; ++j) acc2[nt][j] = 0.0f;

#pragma unroll
  for (int kt = 0; kt < 4; ++kt) {
    int k1 = kt * 32 + hi * 8;
    int k2 = kt * 32 + 16 + hi * 8;
    v8h g0 = *(const v8h*)&s_H[w][rowA][k1];
    v8h g1 = *(const v8h*)&s_H[w][rowA][k2];
    v16h a;
#pragma unroll
    for (int j = 0; j < 8; ++j) { a[j] = g0[j]; a[8 + j] = g1[j]; }
#pragma unroll
    for (int nt = 0; nt < 8; ++nt) {
      v16h bf = *(const v16h*)(w2 + ((size_t)((kt * 8 + nt) * 32 + lane)) * 16);
      acc2[nt] = __builtin_amdgcn_wmma_f32_16x16x32_f16(
          false, a, false, bf, (short)0, acc2[nt], false, false);
    }
  }

#pragma unroll
  for (int nt = 0; nt < 8; ++nt) {
    float bv = b2[nt * 16 + col];
#pragma unroll
    for (int r = 0; r < 8; ++r) acc2[nt][r] += bv;
  }
  float sum[8], sq[8];
#pragma unroll
  for (int r = 0; r < 8; ++r) { sum[r] = 0.0f; sq[r] = 0.0f; }
#pragma unroll
  for (int nt = 0; nt < 8; ++nt)
#pragma unroll
    for (int r = 0; r < 8; ++r) { float y = acc2[nt][r]; sum[r] += y; sq[r] += y * y; }
#pragma unroll
  for (int m = 1; m < 16; m <<= 1) {
#pragma unroll
    for (int r = 0; r < 8; ++r) {
      sum[r] += __shfl_xor(sum[r], m);
      sq[r]  += __shfl_xor(sq[r], m);
    }
  }
#pragma unroll
  for (int nt = 0; nt < 8; ++nt) {
    int n = nt * 16 + col;
    float gv = gln[n], bev = bln[n];
#pragma unroll
    for (int r = 0; r < 8; ++r) {
      float muv = sum[r] * (1.0f / 128.0f);
      float var = sq[r] * (1.0f / 128.0f) - muv * muv;
      float rsv = rsqrtf(var + 1e-5f);
      float yn = (acc2[nt][r] - muv) * rsv * gv + bev;
      size_t gm = (size_t)tile * 16 + hi * 8 + r;
      outV[gm * ND_ + n] = V[gm * ND_ + n] + yn;
    }
  }
}

// ---------------------------------------------------------------------------
extern "C" void kernel_launch(void* const* d_in, const int* in_sizes, int n_in,
                              void* d_out, int out_size, void* d_ws, size_t ws_size,
                              hipStream_t stream) {
  (void)in_sizes; (void)n_in; (void)out_size; (void)ws_size;

  const float* V       = (const float*)d_in[0];
  const float* E       = (const float*)d_in[1];
  const int*   edges   = (const int*)d_in[2];
  const float* fe_w1   = (const float*)d_in[3];
  const float* fe_b1   = (const float*)d_in[4];
  const float* fe_w2   = (const float*)d_in[5];
  const float* fe_b2   = (const float*)d_in[6];
  const float* fe_g    = (const float*)d_in[7];
  const float* fe_beta = (const float*)d_in[8];
  const float* fn_w1   = (const float*)d_in[9];
  const float* fn_b1   = (const float*)d_in[10];
  const float* fn_w2   = (const float*)d_in[11];
  const float* fn_b2   = (const float*)d_in[12];
  const float* fn_g    = (const float*)d_in[13];
  const float* fn_beta = (const float*)d_in[14];

  float* outV = (float*)d_out;
  float* outE = outV + (size_t)B_ * N_ * ND_;

  char* ws = (char*)d_ws;
  float*     aggSum = (float*)(ws + AGG_OFF);
  float*     cntv   = (float*)(ws + CNT_OFF);
  _Float16*  w1e    = (_Float16*)(ws + W1E_OFF);
  _Float16*  w2e    = (_Float16*)(ws + W2E_OFF);
  _Float16*  w1n    = (_Float16*)(ws + W1N_OFF);
  _Float16*  w2n    = (_Float16*)(ws + W2N_OFF);
  _Float16*  Vh     = (_Float16*)(ws + VH_OFF);

  // Pack weights to WMMA B-fragment fp16 layout (cheap; L2-resident after).
  pack_w_kernel<<<(384 * 128 + 255) / 256, 256, 0, stream>>>(fe_w1, w1e, 384);
  pack_w_kernel<<<(128 * 128 + 255) / 256, 256, 0, stream>>>(fe_w2, w2e, 128);
  pack_w_kernel<<<(256 * 128 + 255) / 256, 256, 0, stream>>>(fn_w1, w1n, 256);
  pack_w_kernel<<<(128 * 128 + 255) / 256, 256, 0, stream>>>(fn_w2, w2n, 128);

  // Pre-convert V to fp16 (gathered 64x per row on average; convert once).
  int vn = B_ * N_ * ND_;
  conv_v_kernel<<<(vn + 255) / 256, 256, 0, stream>>>(V, Vh, vn);

  // Zero scatter sums + counts (contiguous in workspace).
  int zn = B_ * N_ * ED_ + B_ * N_;
  zero_kernel<<<(zn + 255) / 256, 256, 0, stream>>>(aggSum, zn);

  edge_kernel<<<NBLK_E, WPB_E * 32, 0, stream>>>(
      Vh, E, edges, w1e, w2e, fe_b1, fe_b2, fe_g, fe_beta, aggSum, cntv, outE);

  node_kernel<<<NODE_TILES / WPB_N, WPB_N * 32, 0, stream>>>(
      V, Vh, aggSum, cntv, w1n, w2n, fn_b1, fn_b2, fn_g, fn_beta, outV);
}
